// Router_78245714198528
// MI455X (gfx1250) — matrix operand order repaired and checked by reading, not compiled, not run on hardware
//
#include <hip/hip_runtime.h>
#include <stdint.h>

// ---------------------------------------------------------------------------
// MoE router for MI455X (gfx1250, wave32).
// logits = x(32768x4096, fp32) @ W(4096x64, fp32) -> softmax -> top8 -> softmax
// Bandwidth-bound: 512MB of x @ 23.3TB/s ~ 22us. AI = 32 flop/B -> needs
// ~745 TFLOP/s matmul rate; use bf16 WMMA (v_wmma_f32_16x16x32_bf16) with
// on-the-fly fp32->bf16 conversion (v_cvt_pk_bf16_f32 co-executes with XDL).
// x and the pre-swizzled bf16 W stream through LDS via the Tensor Data Mover
// (tensor_load_to_lds), 4-deep ring buffer, prefetch distance 3 chunks.
// ---------------------------------------------------------------------------

typedef __attribute__((ext_vector_type(16))) __bf16   v16bf;
typedef __attribute__((ext_vector_type(8)))  __bf16   v8bf;
typedef __attribute__((ext_vector_type(8)))  float    v8f;
typedef __attribute__((ext_vector_type(4)))  float    v4f;
typedef __attribute__((ext_vector_type(4)))  uint32_t u32x4;
typedef __attribute__((ext_vector_type(8)))  uint32_t u32x8;

#define BT        32768
#define DDIM      4096
#define EEXP      64
#define TOPK      8
#define KC        32                 // K depth per chunk (one bf16 WMMA)
#define NCHUNK    (DDIM / KC)        // 128
#define TOK_BLK   64                 // tokens per block (4 waves x 16)
#define BUFS      4                  // LDS ring depth
#define PF        3                  // chunks prefetched ahead

// ---- LDS layout (all dynamic shared memory; byte offsets) ----
#define XROW_W     36                                  // 32 data + 4 pad dwords
#define XBUF_BYTES (TOK_BLK * XROW_W * 4)              // 9216
#define BFRAG_STR  48                                  // 32 data + 16 pad bytes
#define BBUF_BYTES (4 * 32 * BFRAG_STR)                // 6144
#define BBUF_BASE  (BUFS * XBUF_BYTES)                 // 36864
#define EPOFS      (BBUF_BASE + BUFS * BBUF_BYTES)     // 61440
#define EP_W       68                                  // padded row, conflict-free
#define SMEM_BYTES (EPOFS + TOK_BLK * EP_W * 4)        // 78848 bytes

// ---------------------------------------------------------------------------
// Kernel 1: swizzle W (4096x64 fp32, row-major) into bf16 B-fragments.
// Per 32-deep chunk c, expert-group g (16 experts), lane l, element j:
//   B[lane=l][j] = W[k = c*32 + 16*(l/16) + j][e = g*16 + (l%16)]
// (B matrix: one row striped across lanes within one VGPR; 2 halves/VGPR.)
// ---------------------------------------------------------------------------
__global__ void prep_B(const float* __restrict__ w, __bf16* __restrict__ bswz) {
  int idx = blockIdx.x * blockDim.x + threadIdx.x;   // 128*4*32*16 = 262144
  int j = idx & 15;
  int l = (idx >> 4) & 31;
  int g = (idx >> 9) & 3;
  int c = idx >> 11;
  int n = l & 15, h = l >> 4;
  int k = c * KC + h * 16 + j;
  bswz[idx] = (__bf16)w[k * EEXP + g * 16 + n];
}

// ---------------------------------------------------------------------------
// TDM issue helper: builds D# group0 in SGPRs, group1 passed in.
// ---------------------------------------------------------------------------
__device__ __forceinline__ void tdm_issue(uint32_t lds_off, uint64_t gaddr,
                                          u32x8 g1) {
  u32x4 g0;
  g0.x = 1u;                                            // count=1, user mode
  g0.y = lds_off;                                       // lds_addr
  g0.z = (uint32_t)gaddr;                               // global_addr[31:0]
  g0.w = ((uint32_t)(gaddr >> 32) & 0x01FFFFFFu)        // global_addr[56:32]
         | 0x80000000u;                                 // type=2 ("image")
  asm volatile("tensor_load_to_lds %0, %1" :: "s"(g0), "s"(g1) : "memory");
}

// ---------------------------------------------------------------------------
// Kernel 2: fused GEMM (bf16 WMMA) + softmax + top-8 + renorm softmax.
// ---------------------------------------------------------------------------
__global__ __launch_bounds__(128) void router_fused(
    const float* __restrict__ x, const __bf16* __restrict__ bswz,
    float* __restrict__ out_w, int* __restrict__ out_i) {
  extern __shared__ char smem[];
  float* smemf = (float*)smem;
  const uint32_t lds_base = __builtin_amdgcn_groupstaticsize();  // dyn LDS base

  const int tid  = threadIdx.x;
  const int wave = tid >> 5;
  const int lane = tid & 31;
  const int n    = lane & 15;      // M (A) / N (B) index within half-wave
  const int h    = lane >> 4;      // lane half
  const uint64_t tokBase = (uint64_t)blockIdx.x * TOK_BLK;

  // --- TDM descriptor group1 for the x tile: 2D, 64 rows x 32 dwords,
  //     row stride 4096 dwords, LDS pad 4 dwords per 32 (row stride 36) ---
  const u32x8 g1x = { (2u << 16) | (1u << 20) | (4u << 22) | (3u << 25),
                      (uint32_t)DDIM << 16,        // tensor_dim0[15:0]
                      (uint32_t)TOK_BLK << 16,     // tensor_dim1[15:0]
                      32u << 16,                   // tile_dim0
                      (uint32_t)TOK_BLK,           // tile_dim1
                      (uint32_t)DDIM,              // tensor_dim0_stride[31:0]
                      0u, 0u };
  // --- group1 for B chunk: 1D, 1024 dwords, LDS pad 4 dwords per 8
  //     (each 32B fragment lands on a 48B slot) ---
  const u32x8 g1b = { (2u << 16) | (1u << 20) | (2u << 22) | (3u << 25),
                      1024u << 16,                 // tensor_dim0
                      1u << 16,                    // tensor_dim1 = 1
                      1024u << 16,                 // tile_dim0
                      0u,                          // tile_dim1 unused
                      1024u,                       // tensor_dim0_stride
                      0u, 0u };

  const uint64_t xbase = (uint64_t)(uintptr_t)x + tokBase * (DDIM * 4);
  const uint64_t bbase = (uint64_t)(uintptr_t)bswz;

  if (wave == 0) {                       // prologue: chunks 0..PF-1 -> slots
    #pragma unroll
    for (int i = 0; i < PF; ++i) {
      tdm_issue(lds_base + i * XBUF_BYTES, xbase + (uint64_t)i * (KC * 4), g1x);
      tdm_issue(lds_base + BBUF_BASE + i * BBUF_BYTES,
                bbase + (uint64_t)i * (KC * EEXP * 2), g1b);
    }
  }

  v8f acc[4] = {v8f{0.f}, v8f{0.f}, v8f{0.f}, v8f{0.f}};

  #pragma unroll 4
  for (int c = 0; c < NCHUNK; ++c) {
    // 2 TDM ops per chunk; allow PF-1 chunks (4 ops) beyond chunk c in flight.
    __builtin_amdgcn_s_wait_tensorcnt(2 * (PF - 1));   // chunk c landed
    __syncthreads();                                   // publish + retire reads
    if (wave == 0) {
      // Always issue into slot (c+PF)%BUFS to keep the tensorcnt invariant;
      // past the end, clamp the source chunk (dummy load, slot never read).
      int pc = c + PF;
      const int slot = pc & (BUFS - 1);
      if (pc > NCHUNK - 1) pc = NCHUNK - 1;
      tdm_issue(lds_base + slot * XBUF_BYTES,
                xbase + (uint64_t)pc * (KC * 4), g1x);
      tdm_issue(lds_base + BBUF_BASE + slot * BBUF_BYTES,
                bbase + (uint64_t)pc * (KC * EEXP * 2), g1b);
    }
    const int cur = c & (BUFS - 1);      // constant after unroll-4

    // A fragment: 16 tokens x 32 K, fp32 from LDS -> bf16 regs.
    // lane holds row (wave*16+n); K = 16*h + j.
    const float* xb = smemf + (cur * XBUF_BYTES) / 4 + (wave * 16 + n) * XROW_W;
    v4f u0 = *(const v4f*)(xb + 16 * h);
    v4f u1 = *(const v4f*)(xb + 16 * h + 4);
    v4f u2 = *(const v4f*)(xb + 16 * h + 8);
    v4f u3 = *(const v4f*)(xb + 16 * h + 12);
    v16bf A;
    #pragma unroll
    for (int i = 0; i < 4; ++i) {
      A[i]      = (__bf16)u0[i];
      A[4 + i]  = (__bf16)u1[i];
      A[8 + i]  = (__bf16)u2[i];
      A[12 + i] = (__bf16)u3[i];
    }

    // Load ALL 4 expert-group B fragments first (distinct regs), then issue
    // 4 independent WMMAs so the XDL pipe isn't gated on per-load dscnt=0.
    const char* bb = smem + BBUF_BASE + cur * BBUF_BYTES + lane * BFRAG_STR;
    v16bf Bf[4];
    #pragma unroll
    for (int g = 0; g < 4; ++g) {
      v8bf blo = *(const v8bf*)(bb + g * 32 * BFRAG_STR);
      v8bf bhi = *(const v8bf*)(bb + g * 32 * BFRAG_STR + 16);
      Bf[g] = __builtin_shufflevector(blo, bhi, 0, 1, 2, 3, 4, 5, 6, 7,
                                      8, 9, 10, 11, 12, 13, 14, 15);
    }
    #pragma unroll
    for (int g = 0; g < 4; ++g)
      acc[g] = __builtin_amdgcn_wmma_f32_16x16x32_bf16(
          false, A, false, Bf[g], (short)0, acc[g], false, false);
  }

  // ---- dump logits: C layout VGPR r -> M = r + 8h, N = g*16 + n ----
  float* ep = smemf + EPOFS / 4;
  {
    const int row0 = wave * 16 + 8 * h;
    #pragma unroll
    for (int g = 0; g < 4; ++g) {
      #pragma unroll
      for (int r = 0; r < 8; ++r)
        ep[(row0 + r) * EP_W + g * 16 + n] = acc[g][r];
    }
  }
  __syncthreads();

  // ---- epilogue: 1 token per thread for tid < 64 ----
  if (tid < TOK_BLK) {
    const float* row = ep + tid * EP_W;
    float v[64];
    float m = -3.4e38f;
    #pragma unroll
    for (int j = 0; j < 64; ++j) { v[j] = row[j]; m = fmaxf(m, v[j]); }
    float denom = 0.f;
    #pragma unroll
    for (int j = 0; j < 64; ++j) { v[j] = __expf(v[j] - m); denom += v[j]; }

    // stable top-8 on (value desc, index asc) without dynamic reg indexing
    float pv = 3.4e38f; int pj = -1;
    float wv[TOPK]; int wi[TOPK];
    #pragma unroll
    for (int k = 0; k < TOPK; ++k) {
      float bv = -1.f; int bj = 64;
      #pragma unroll
      for (int j = 0; j < 64; ++j) {
        bool lt_prev = (v[j] < pv) || ((v[j] == pv) && (j > pj));
        bool gt_best = (v[j] > bv) || ((v[j] == bv) && (j < bj));
        if (lt_prev && gt_best) { bv = v[j]; bj = j; }
      }
      wv[k] = bv; wi[k] = bj; pv = bv; pj = bj;
    }

    // weights are softmax probs: wv/denom; renormalize with a second softmax
    const float inv_denom = 1.f / denom;
    const float w0 = wv[0] * inv_denom;                 // max (sorted desc)
    float e[TOPK]; float s = 0.f;
    #pragma unroll
    for (int k = 0; k < TOPK; ++k) { e[k] = __expf(wv[k] * inv_denom - w0); s += e[k]; }
    const float is = 1.f / s;

    const uint64_t gtok = tokBase + (uint64_t)tid;
    #pragma unroll
    for (int k = 0; k < TOPK; ++k) {
      out_w[gtok * TOPK + k] = e[k] * is;
      out_i[gtok * TOPK + k] = wi[k];
    }
  }
}

// ---------------------------------------------------------------------------
extern "C" void kernel_launch(void* const* d_in, const int* in_sizes, int n_in,
                              void* d_out, int out_size, void* d_ws, size_t ws_size,
                              hipStream_t stream) {
  (void)in_sizes; (void)n_in; (void)out_size; (void)ws_size;
  const float* x   = (const float*)d_in[0];
  const float* w   = (const float*)d_in[1];
  __bf16*      bsw = (__bf16*)d_ws;                  // 512 KB swizzled bf16 W
  float*       out_w = (float*)d_out;                // [32768][8] fp32
  int*         out_i = (int*)d_out + BT * TOPK;      // [32768][8] int32

  prep_B<<<(NCHUNK * 4 * 32 * 16) / 256, 256, 0, stream>>>(w, bsw);
  router_fused<<<BT / TOK_BLK, 128, SMEM_BYTES, stream>>>(x, bsw, out_w, out_i);
}
